// LRU_new_61349312856305
// MI455X (gfx1250) — matrix operand order repaired and checked by reading, not compiled, not run on hardware
//
#include <hip/hip_runtime.h>
#include <hip/hip_bf16.h>

// ---------------------------------------------------------------------------
// LRU forward (diagonalized). The Euler step diagonalizes to
//   z_{t+1} = a (.) z_t + h * f_t,   f_t = u_t @ Bc^T,   xi_t = Re/Im(z) @ Gcat^T
// Stages:
//   k_prep        : a and a^128 per mode (7 complex squarings)
//   k_pack_Q      : QTc = Pinv transposed (re|im concat)         (for z0)
//   k_pack_B      : WMMA B fragments of [Bre|Bim]^T (bf16 hi/lo split)
//   k_pack_G      : WMMA B fragments of Gcat^T = [2Pre | -2Pim]^T (hi/lo)
//   k_z0          : z0 = xi0 @ Q^T (fp32)
//   k_fgemm       : f = u @ [Bre|Bim]^T     (WMMA bf16, 3-product split)
//   k_scan_chunks : per-chunk forcing aggregate g_c (256 blocks, parallel)
//   k_scan_carry  : chunk-boundary states via z' = a^L z + g  (tiny)
//   k_scan_emit   : re-run chunks from boundaries, write zlog (256 blocks)
//   k_outgemm     : out = z @ Gcat^T        (WMMA bf16, 3-product split)
//   k_copy0       : out[:,0,:] = xi0        (reference skips xi_1)
// ---------------------------------------------------------------------------

#define N_STATE 256
#define TWO_N   512
#define N_IN    128
#define BATCH   16
#define T_STEPS 2048
#define NCHUNK  16
#define CLEN    128   // T_STEPS / NCHUNK, == 2^7
#define H_STEP  0.01f
#define EPS_LRU 0.001f

typedef __attribute__((ext_vector_type(16))) __bf16 v16bf;
typedef __attribute__((ext_vector_type(8)))  float  v8f;
typedef __attribute__((ext_vector_type(8)))  int    v8i_;

// ---- bf16 helpers ---------------------------------------------------------
__device__ __forceinline__ unsigned f2bf_rne(float f) {
  unsigned u = __builtin_bit_cast(unsigned, f);
  return (u + 0x7FFFu + ((u >> 16) & 1u)) >> 16;
}
__device__ __forceinline__ float bf2f(unsigned h) {
  return __builtin_bit_cast(float, h << 16);
}
__device__ __forceinline__ void bf_split(float x, unsigned& h, unsigned& l) {
  h = f2bf_rne(x);
  float hf = bf2f(h);
  l = f2bf_rne(x - hf);
}
__device__ __forceinline__ void pack_pair_split(float x0, float x1,
                                                unsigned& dh, unsigned& dl) {
  unsigned h0, l0, h1, l1;
  bf_split(x0, h0, l0);
  bf_split(x1, h1, l1);
  dh = h0 | (h1 << 16);
  dl = l0 | (l1 << 16);
}

// A-fragment K mapping (16-bit A 16x32, CDNA5 ISA §7.12.2)
__device__ __forceinline__ int a_kbase(int r, int half) {
  return (r < 4 ? 2 * r : 8 + 2 * r) + 8 * half;  // pair (k, k+1)
}

__device__ __forceinline__ v8f wmma_bf16(v8i_ a, v8i_ b, v8f c) {
  return __builtin_amdgcn_wmma_f32_16x16x32_bf16(
      false, __builtin_bit_cast(v16bf, a),
      false, __builtin_bit_cast(v16bf, b),
      (short)0, c, false, false);
}

// ---- small prep kernels ---------------------------------------------------
__global__ void k_prep(const float* __restrict__ alpha,
                       const float* __restrict__ theta,
                       float* __restrict__ are, float* __restrict__ aim,
                       float* __restrict__ aLre, float* __restrict__ aLim) {
  int k = threadIdx.x;  // 256 threads
  float ar = 1.0f - H_STEP * (expf(alpha[k]) + EPS_LRU);
  float ai = H_STEP * theta[k];
  are[k] = ar;
  aim[k] = ai;
  // a^CLEN via log2(CLEN)=7 complex squarings
  float cr = ar, ci = ai;
#pragma unroll
  for (int s = 0; s < 7; s++) {
    float nr = cr * cr - ci * ci;
    float ni = 2.0f * cr * ci;
    cr = nr;
    ci = ni;
  }
  aLre[k] = cr;
  aLim[k] = ci;
}

// QTc[j*512 + kk]: kk<256 -> Pinv_re[kk][j], else Pinv_im[kk-256][j]
__global__ void k_pack_Q(const float* __restrict__ pinv_re,
                         const float* __restrict__ pinv_im,
                         float* __restrict__ QTc) {
  int idx = blockIdx.x * 256 + threadIdx.x;  // 512*512
  int j = idx >> 9, kk = idx & 511;
  QTc[idx] = (kk < N_STATE) ? pinv_re[kk * TWO_N + j]
                            : pinv_im[(kk - N_STATE) * TWO_N + j];
}

// B fragments for f-GEMM: K=128, N=512.  BB[k][col] = (col<256 ? Bre[col][k]
// : Bim[col-256][k]).  Layout [(kt*32+nt)*32+lane]*8 dwords, kt<4.
__global__ void k_pack_B(const float* __restrict__ Bre,
                         const float* __restrict__ Bim,
                         unsigned* __restrict__ bhi, unsigned* __restrict__ blo) {
  int idx = blockIdx.x * 128 + threadIdx.x;  // 4*32*32
  int lane = idx & 31, nt = (idx >> 5) & 31, kt = idx >> 10;
  int n = lane & 15, half = lane >> 4;
  int col = nt * 16 + n;
  const float* src = (col < N_STATE) ? (Bre + col * N_IN)
                                     : (Bim + (col - N_STATE) * N_IN);
#pragma unroll
  for (int r = 0; r < 8; r++) {
    int k = kt * 32 + a_kbase(r, half);  // B 32x16: same K order as A fragment
    unsigned dh, dl;
    pack_pair_split(src[k], src[k + 1], dh, dl);
    bhi[idx * 8 + r] = dh;
    blo[idx * 8 + r] = dl;
  }
}

// B fragments for out-GEMM: K=512, N=512.
// Gcat^T[kk][j] = kk<256 ? 2*P_re[j][kk] : -2*P_im[j][kk-256].  kt<16.
__global__ void k_pack_G(const float* __restrict__ P_re,
                         const float* __restrict__ P_im,
                         unsigned* __restrict__ ghi, unsigned* __restrict__ glo) {
  int idx = blockIdx.x * 128 + threadIdx.x;  // 16*32*32
  int lane = idx & 31, nt = (idx >> 5) & 31, kt = idx >> 10;
  int n = lane & 15, half = lane >> 4;
  int j = nt * 16 + n;
#pragma unroll
  for (int r = 0; r < 8; r++) {
    int kk = kt * 32 + a_kbase(r, half);
    float x0 = (kk < N_STATE) ? 2.0f * P_re[j * TWO_N + kk]
                              : -2.0f * P_im[j * TWO_N + (kk - N_STATE)];
    int kk1 = kk + 1;
    float x1 = (kk1 < N_STATE) ? 2.0f * P_re[j * TWO_N + kk1]
                               : -2.0f * P_im[j * TWO_N + (kk1 - N_STATE)];
    unsigned dh, dl;
    pack_pair_split(x0, x1, dh, dl);
    ghi[idx * 8 + r] = dh;
    glo[idx * 8 + r] = dl;
  }
}

// z0[b*512+kk] = sum_j xi0[b][j] * QTc[j][kk]   (fp32, tiny)
__global__ __launch_bounds__(256) void k_z0(const float* __restrict__ xi0,
                                            const float* __restrict__ QTc,
                                            float* __restrict__ z0) {
  __shared__ float xrow[TWO_N];
  int b = blockIdx.x >> 1;
  int kk = (blockIdx.x & 1) * 256 + threadIdx.x;
  for (int j = threadIdx.x; j < TWO_N; j += 256) xrow[j] = xi0[b * TWO_N + j];
  __syncthreads();
  float acc = 0.0f;
  for (int j = 0; j < TWO_N; j++) acc = fmaf(xrow[j], QTc[j * TWO_N + kk], acc);
  z0[b * TWO_N + kk] = acc;
}

// ---- f = u @ [Bre|Bim]^T : [32768,128] x [128,512], WMMA bf16 hi/lo -------
__global__ __launch_bounds__(256, 1) void k_fgemm(
    const float* __restrict__ u, const unsigned* __restrict__ bhi,
    const unsigned* __restrict__ blo, float* __restrict__ f) {
  int wave = threadIdx.x >> 5, lane = threadIdx.x & 31;
  int mt = blockIdx.x * 8 + wave;           // 2048 M-tiles of 16 rows
  int m = lane & 15, half = lane >> 4;
  const float* urow = u + (size_t)(mt * 16 + m) * N_IN;

  v8i_ Ah[4], Al[4];
#pragma unroll
  for (int kt = 0; kt < 4; kt++) {
#pragma unroll
    for (int r = 0; r < 8; r++) {
      int k = kt * 32 + a_kbase(r, half);
      unsigned dh, dl;
      pack_pair_split(urow[k], urow[k + 1], dh, dl);
      Ah[kt][r] = (int)dh;
      Al[kt][r] = (int)dl;
    }
  }

  for (int nt = 0; nt < 32; nt++) {
    if (nt + 1 < 32)
      __builtin_prefetch(bhi + ((size_t)(nt + 1) * 32 + lane) * 8, 0, 0);
    v8f c = {};
#pragma unroll
    for (int kt = 0; kt < 4; kt++) {
      size_t fo = ((size_t)(kt * 32 + nt) * 32 + lane) * 8;
      v8i_ Bh = *(const v8i_*)(bhi + fo);
      v8i_ Bl = *(const v8i_*)(blo + fo);
      c = wmma_bf16(Ah[kt], Bh, c);
      c = wmma_bf16(Ah[kt], Bl, c);
      c = wmma_bf16(Al[kt], Bh, c);
    }
    float* frow = f + (size_t)(mt * 16 + half * 8) * TWO_N + nt * 16 + m;
#pragma unroll
    for (int r = 0; r < 8; r++) frow[(size_t)r * TWO_N] = c[r];
  }
}

// ---- chunked diagonal scan ------------------------------------------------
// Pass A: per-chunk forcing aggregate g_c = sum_j a^(CLEN-1-j) * h * f_j.
// 256 blocks (b,c) x 256 lanes (k); inner loop unrolled x8 so the
// independent f-loads batch into clauses ahead of the FMA carry chain.
__global__ __launch_bounds__(256) void k_scan_chunks(
    const float* __restrict__ are_, const float* __restrict__ aim_,
    const float* __restrict__ f, float* __restrict__ g) {
  int b = blockIdx.x >> 4, c = blockIdx.x & 15, k = threadIdx.x;
  float ar = are_[k], ai = aim_[k];
  const float* fb = f + ((size_t)b * T_STEPS + c * CLEN) * TWO_N;
  float gr = 0.0f, gi = 0.0f;
  for (int j0 = 0; j0 < CLEN; j0 += 8) {
    float fr[8], fi[8];
#pragma unroll
    for (int j = 0; j < 8; j++) {
      fr[j] = fb[(size_t)(j0 + j) * TWO_N + k];
      fi[j] = fb[(size_t)(j0 + j) * TWO_N + N_STATE + k];
    }
#pragma unroll
    for (int j = 0; j < 8; j++) {
      float nr = fmaf(ar, gr, fmaf(-ai, gi, H_STEP * fr[j]));
      float ni = fmaf(ar, gi, fmaf(ai, gr, H_STEP * fi[j]));
      gr = nr;
      gi = ni;
    }
  }
  size_t o = (size_t)blockIdx.x * TWO_N;  // index (b*16+c)
  g[o + k] = gr;
  g[o + N_STATE + k] = gi;
}

// Pass B1: chunk-boundary states  z_{(c+1)L} = a^L z_{cL} + g_c  (tiny)
__global__ __launch_bounds__(256) void k_scan_carry(
    const float* __restrict__ z0, const float* __restrict__ aLre_,
    const float* __restrict__ aLim_, const float* __restrict__ g,
    float* __restrict__ zstart) {
  int b = blockIdx.x, k = threadIdx.x;
  float ar = aLre_[k], ai = aLim_[k];
  float zr = z0[b * TWO_N + k];
  float zi = z0[b * TWO_N + N_STATE + k];
  for (int c = 0; c < NCHUNK; c++) {
    size_t o = (size_t)(b * NCHUNK + c) * TWO_N;
    zstart[o + k] = zr;
    zstart[o + N_STATE + k] = zi;
    float gr = g[o + k], gi = g[o + N_STATE + k];
    float nr = fmaf(ar, zr, fmaf(-ai, zi, gr));
    float ni = fmaf(ar, zi, fmaf(ai, zr, gi));
    zr = nr;
    zi = ni;
  }
}

// Pass B2: re-run each chunk from its boundary state; store z_{t+1} at row t.
__global__ __launch_bounds__(256) void k_scan_emit(
    const float* __restrict__ zstart, const float* __restrict__ are_,
    const float* __restrict__ aim_, const float* __restrict__ f,
    float* __restrict__ zlog) {
  int b = blockIdx.x >> 4, c = blockIdx.x & 15, k = threadIdx.x;
  float ar = are_[k], ai = aim_[k];
  size_t o = (size_t)blockIdx.x * TWO_N;
  float zr = zstart[o + k], zi = zstart[o + N_STATE + k];
  const float* fb = f + ((size_t)b * T_STEPS + c * CLEN) * TWO_N;
  float* zb = zlog + ((size_t)b * T_STEPS + c * CLEN) * TWO_N;
  for (int j0 = 0; j0 < CLEN; j0 += 8) {
    float fr[8], fi[8];
#pragma unroll
    for (int j = 0; j < 8; j++) {
      fr[j] = fb[(size_t)(j0 + j) * TWO_N + k];
      fi[j] = fb[(size_t)(j0 + j) * TWO_N + N_STATE + k];
    }
#pragma unroll
    for (int j = 0; j < 8; j++) {
      float nr = fmaf(ar, zr, fmaf(-ai, zi, H_STEP * fr[j]));
      float ni = fmaf(ar, zi, fmaf(ai, zr, H_STEP * fi[j]));
      zr = nr;
      zi = ni;
      zb[(size_t)(j0 + j) * TWO_N + k] = zr;
      zb[(size_t)(j0 + j) * TWO_N + N_STATE + k] = zi;
    }
  }
}

// ---- out = z @ Gcat^T : [32768,512] x [512,512], WMMA bf16 hi/lo ----------
__global__ __launch_bounds__(256, 1) void k_outgemm(
    const float* __restrict__ z, const unsigned* __restrict__ ghi,
    const unsigned* __restrict__ glo, float* __restrict__ out) {
  int wave = threadIdx.x >> 5, lane = threadIdx.x & 31;
  int mt = blockIdx.x * 8 + wave;
  int m = lane & 15, half = lane >> 4;
  const float* zrow = z + (size_t)(mt * 16 + m) * TWO_N;

  v8i_ Ah[16], Al[16];
#pragma unroll
  for (int kt = 0; kt < 16; kt++) {
#pragma unroll
    for (int r = 0; r < 8; r++) {
      int k = kt * 32 + a_kbase(r, half);
      unsigned dh, dl;
      pack_pair_split(zrow[k], zrow[k + 1], dh, dl);
      Ah[kt][r] = (int)dh;
      Al[kt][r] = (int)dl;
    }
  }

  for (int nt = 0; nt < 32; nt++) {
    if (nt + 1 < 32)
      __builtin_prefetch(ghi + ((size_t)(nt + 1) * 32 + lane) * 8, 0, 0);
    v8f c = {};
#pragma unroll
    for (int kt = 0; kt < 16; kt++) {
      size_t fo = ((size_t)(kt * 32 + nt) * 32 + lane) * 8;
      v8i_ Bh = *(const v8i_*)(ghi + fo);
      v8i_ Bl = *(const v8i_*)(glo + fo);
      c = wmma_bf16(Ah[kt], Bh, c);
      c = wmma_bf16(Ah[kt], Bl, c);
      c = wmma_bf16(Al[kt], Bh, c);
    }
    float* orow = out + (size_t)(mt * 16 + half * 8) * TWO_N + nt * 16 + m;
#pragma unroll
    for (int r = 0; r < 8; r++) orow[(size_t)r * TWO_N] = c[r];
  }
}

__global__ void k_copy0(const float* __restrict__ xi0, float* __restrict__ out) {
  int idx = blockIdx.x * 256 + threadIdx.x;  // 16*512
  int b = idx >> 9, j = idx & 511;
  out[(size_t)b * T_STEPS * TWO_N + j] = xi0[idx];
}

// ---------------------------------------------------------------------------
extern "C" void kernel_launch(void* const* d_in, const int* in_sizes, int n_in,
                              void* d_out, int out_size, void* d_ws,
                              size_t ws_size, hipStream_t stream) {
  const float* xi_init = (const float*)d_in[0];   // [16,1,512]
  const float* u_log   = (const float*)d_in[1];   // [16,2048,128]
  const float* alpha   = (const float*)d_in[2];   // [256]
  const float* theta   = (const float*)d_in[3];   // [256]
  const float* B_re    = (const float*)d_in[4];   // [256,128]
  const float* B_im    = (const float*)d_in[5];   // [256,128]
  const float* P_re    = (const float*)d_in[6];   // [512,512]
  const float* P_im    = (const float*)d_in[7];   // [512,512]
  const float* Pinv_re = (const float*)d_in[8];   // [512,512]
  const float* Pinv_im = (const float*)d_in[9];   // [512,512]
  float* out = (float*)d_out;                     // [16,2048,512]

  char* ws = (char*)d_ws;
  size_t off = 0;
  auto take = [&](size_t bytes) -> void* {
    void* p = ws + off;
    off = (off + bytes + 255) & ~(size_t)255;
    return p;
  };
  float*    are    = (float*)take(N_STATE * 4);
  float*    aim    = (float*)take(N_STATE * 4);
  float*    aLre   = (float*)take(N_STATE * 4);
  float*    aLim   = (float*)take(N_STATE * 4);
  float*    z0     = (float*)take(BATCH * TWO_N * 4);
  float*    QTc    = (float*)take((size_t)TWO_N * TWO_N * 4);
  unsigned* bhi    = (unsigned*)take((size_t)4 * 32 * 32 * 8 * 4);
  unsigned* blo    = (unsigned*)take((size_t)4 * 32 * 32 * 8 * 4);
  unsigned* ghi    = (unsigned*)take((size_t)16 * 32 * 32 * 8 * 4);
  unsigned* glo    = (unsigned*)take((size_t)16 * 32 * 32 * 8 * 4);
  float*    gbuf   = (float*)take((size_t)BATCH * NCHUNK * TWO_N * 4);
  float*    zstart = (float*)take((size_t)BATCH * NCHUNK * TWO_N * 4);
  float*    f      = (float*)take((size_t)BATCH * T_STEPS * TWO_N * 4);  // 64MB
  float*    zlog   = (float*)take((size_t)BATCH * T_STEPS * TWO_N * 4);  // 64MB

  k_prep<<<1, 256, 0, stream>>>(alpha, theta, are, aim, aLre, aLim);
  k_pack_Q<<<(TWO_N * TWO_N) / 256, 256, 0, stream>>>(Pinv_re, Pinv_im, QTc);
  k_pack_B<<<(4 * 32 * 32) / 128, 128, 0, stream>>>(B_re, B_im, bhi, blo);
  k_pack_G<<<(16 * 32 * 32) / 128, 128, 0, stream>>>(P_re, P_im, ghi, glo);
  k_z0<<<BATCH * 2, 256, 0, stream>>>(xi_init, QTc, z0);
  k_fgemm<<<(BATCH * T_STEPS / 16) / 8, 256, 0, stream>>>(u_log, bhi, blo, f);
  k_scan_chunks<<<BATCH * NCHUNK, 256, 0, stream>>>(are, aim, f, gbuf);
  k_scan_carry<<<BATCH, 256, 0, stream>>>(z0, aLre, aLim, gbuf, zstart);
  k_scan_emit<<<BATCH * NCHUNK, 256, 0, stream>>>(zstart, are, aim, f, zlog);
  k_outgemm<<<(BATCH * T_STEPS / 16) / 8, 256, 0, stream>>>(zlog, ghi, glo, out);
  k_copy0<<<(BATCH * TWO_N) / 256, 256, 0, stream>>>(xi_init, out);
}
